// ClipOTLoss_15693810500085
// MI455X (gfx1250) — matrix operand order repaired and checked by reading, not compiled, not run on hardware
//
#include <hip/hip_runtime.h>
#include <hip/hip_bf16.h>
#include <math.h>

// ---------------------------------------------------------------------------
// ClipOTLoss for MI455X (gfx1250, wave32, WMMA, async-to-LDS)
//
//   loss = -mean_b sum_k softcode[b,k] * log_softmax(logits)[b,k]
//   softcode = sinkhorn(exp((feat @ norm(proto)^T / T) / EPS))
//
// Pipeline:
//   1. convert features -> bf16            (elementwise)
//   2. L2-normalize prototypes -> bf16     (block per row)
//   3. WMMA bf16 GEMM, double-buffered LDS filled by
//      global_load_async_to_lds_b128, fused exp epilogue -> Q0 [B,K] f32
//   4. Sinkhorn as row/col factor updates (alpha[K], beta[B]) over Q0
//   5. fused log_softmax + weighted-dot loss, atomicAdd into scalar out
// ---------------------------------------------------------------------------

typedef __attribute__((ext_vector_type(16))) __bf16 v16bf;
typedef __attribute__((ext_vector_type(8)))  float  v8f;

#define TILE_M 128
#define TILE_N 128
#define TILE_K 32
#define LDS_STRIDE 40   // bf16 elems per LDS row: 32 + 8 pad -> 80B, 16B aligned rows

#define TEMPERATURE 0.01f
#define EPSILON     0.7f
#define TINY        1e-8f
#define NUM_ITERS   3
#define KMAX        4096   // K of this problem (loss kernel LDS staging)

__device__ __forceinline__ unsigned short f32_to_bf16(float x) {
    unsigned u = __float_as_uint(x);
    u += 0x7FFFu + ((u >> 16) & 1u);   // round to nearest even
    return (unsigned short)(u >> 16);
}

// CDNA5 async global->LDS copy, 16B per lane (ASYNCcnt-tracked, GV mode).
// ldsOff = low 32 bits of the generic pointer (== LDS byte offset per ISA 10.2).
__device__ __forceinline__ void async_copy_b128(unsigned ldsOff,
                                                const unsigned short* gptr) {
    asm volatile("global_load_async_to_lds_b128 %0, %1, off"
                 :: "v"(ldsOff), "v"(gptr) : "memory");
}
__device__ __forceinline__ void wait_async0() {
    asm volatile("s_wait_asynccnt 0x0" ::: "memory");
}

__device__ __forceinline__ float blockReduceSum(float v, float* red) {
    int t = threadIdx.x;
    red[t] = v; __syncthreads();
    for (int s = 128; s > 0; s >>= 1) { if (t < s) red[t] += red[t + s]; __syncthreads(); }
    float r = red[0]; __syncthreads();
    return r;
}

__device__ __forceinline__ float blockReduceMax(float v, float* red) {
    int t = threadIdx.x;
    red[t] = v; __syncthreads();
    for (int s = 128; s > 0; s >>= 1) { if (t < s) red[t] = fmaxf(red[t], red[t + s]); __syncthreads(); }
    float r = red[0]; __syncthreads();
    return r;
}

// ---------------------------------------------------------------------------
// 0. init: alpha = 1, out = 0
__global__ void init_kernel(float* alpha, float* out, int K) {
    int i = blockIdx.x * blockDim.x + threadIdx.x;
    if (i < K) alpha[i] = 1.0f;
    if (i == 0) out[0] = 0.0f;
}

// 1. features f32 -> bf16
__global__ void feat_cvt_kernel(const float* __restrict__ src,
                                unsigned short* __restrict__ dst, long n) {
    long i = (long)blockIdx.x * blockDim.x + threadIdx.x;
    long stride = (long)gridDim.x * blockDim.x;
    for (; i < n; i += stride) dst[i] = f32_to_bf16(src[i]);
}

// 2. prototypes: L2-normalize each row, emit bf16. One block per row.
__global__ __launch_bounds__(256)
void proto_norm_kernel(const float* __restrict__ proto,
                       unsigned short* __restrict__ dst, int D) {
    __shared__ float red[256];
    const float* row = proto + (size_t)blockIdx.x * D;
    float ss = 0.0f;
    for (int i = threadIdx.x; i < D; i += 256) { float v = row[i]; ss += v * v; }
    ss = blockReduceSum(ss, red);
    float inv = rsqrtf(ss);
    unsigned short* out = dst + (size_t)blockIdx.x * D;
    for (int i = threadIdx.x; i < D; i += 256) out[i] = f32_to_bf16(row[i] * inv);
}

// ---------------------------------------------------------------------------
// 3. WMMA GEMM with fused exp epilogue:
//    Q0[b,k] = exp( (featbf[b,:] . wbf[k,:]) * scale ),  scale = 1/(T*EPS)
//    Block: 256 thr (8 waves), tile 128x128, K-chunks of 32.
//    Double-buffered LDS filled by async-to-LDS; wave grid 2(m) x 4(n),
//    each wave -> 4x2 accumulators of 16x16.
__global__ __launch_bounds__(256)
void gemm_exp_kernel(const unsigned short* __restrict__ fA,   // [B,D] bf16
                     const unsigned short* __restrict__ fW,   // [K,D] bf16 (normalized)
                     float* __restrict__ Q,                   // [B,K] f32 out
                     int B, int K, int D, float scale) {
    __shared__ unsigned short As[2][TILE_M * LDS_STRIDE];
    __shared__ unsigned short Bs[2][TILE_N * LDS_STRIDE];

    const int t     = threadIdx.x;
    const int wave  = t >> 5;
    const int lane  = t & 31;
    const int lhalf = lane >> 4;     // 0 | 1
    const int lm    = lane & 15;
    const int bBase = blockIdx.y * TILE_M;
    const int kBase = blockIdx.x * TILE_N;
    const int wm    = wave & 1;      // m offset 64*wm
    const int wn    = wave >> 1;     // n offset 32*wn

    // This thread's two staging segments (16B each, per tile, per matrix).
    const int row0 = t >> 2,            cs0 = (t & 3) * 8;
    const int row1 = (t + 256) >> 2,    cs1 = cs0;      // seg t+256: rows 64..127
    const unsigned short* gA0 = fA + (size_t)(bBase + row0) * D + cs0;
    const unsigned short* gA1 = fA + (size_t)(bBase + row1) * D + cs1;
    const unsigned short* gB0 = fW + (size_t)(kBase + row0) * D + cs0;
    const unsigned short* gB1 = fW + (size_t)(kBase + row1) * D + cs1;
    const unsigned ldsA0[2] = { (unsigned)(uintptr_t)&As[0][row0 * LDS_STRIDE + cs0],
                                (unsigned)(uintptr_t)&As[1][row0 * LDS_STRIDE + cs0] };
    const unsigned ldsA1[2] = { (unsigned)(uintptr_t)&As[0][row1 * LDS_STRIDE + cs1],
                                (unsigned)(uintptr_t)&As[1][row1 * LDS_STRIDE + cs1] };
    const unsigned ldsB0[2] = { (unsigned)(uintptr_t)&Bs[0][row0 * LDS_STRIDE + cs0],
                                (unsigned)(uintptr_t)&Bs[1][row0 * LDS_STRIDE + cs0] };
    const unsigned ldsB1[2] = { (unsigned)(uintptr_t)&Bs[0][row1 * LDS_STRIDE + cs1],
                                (unsigned)(uintptr_t)&Bs[1][row1 * LDS_STRIDE + cs1] };

    v8f acc[4][2];
    const v8f vzero = {0.f,0.f,0.f,0.f,0.f,0.f,0.f,0.f};
#pragma unroll
    for (int i = 0; i < 4; ++i)
#pragma unroll
        for (int j = 0; j < 2; ++j) acc[i][j] = vzero;

    union Frag { uint4 q[2]; v16bf v; };

    // Prologue: async-fill buffer 0 with the first K-slice.
    async_copy_b128(ldsA0[0], gA0);
    async_copy_b128(ldsA1[0], gA1);
    async_copy_b128(ldsB0[0], gB0);
    async_copy_b128(ldsB1[0], gB1);

    const int nIter = D / TILE_K;
    for (int it = 0; it < nIter; ++it) {
        const int cur = it & 1;
        wait_async0();      // my fills of buf[cur] have landed in LDS
        __syncthreads();    // everyone's fills landed; prior reads of buf[nxt] done

        if (it + 1 < nIter) {            // async-fill the other buffer
            const int nxt = cur ^ 1;
            const int d1  = (it + 1) * TILE_K;
            async_copy_b128(ldsA0[nxt], gA0 + d1);
            async_copy_b128(ldsA1[nxt], gA1 + d1);
            async_copy_b128(ldsB0[nxt], gB0 + d1);
            async_copy_b128(ldsB1[nxt], gB1 + d1);
        }

        // A fragment (ISA 16-bit A 16x32): lanes 0-15 K=0..7(+16..23),
        // lanes 16-31 K=8..15(+24..31); two 16B LDS reads each.
        Frag aF[4], bF[2];
        const int ak0 = lhalf * 8;
#pragma unroll
        for (int mt = 0; mt < 4; ++mt) {
            const unsigned short* p = &As[cur][(wm * 64 + mt * 16 + lm) * LDS_STRIDE + ak0];
            aF[mt].q[0] = *(const uint4*)(p);
            aF[mt].q[1] = *(const uint4*)(p + 16);
        }
        // B fragment (32x16): lanes 0-15 K=0..7(+16..23), lanes 16-31 K=8..15(+24..31).
        const int bk0 = lhalf * 8;
#pragma unroll
        for (int nt = 0; nt < 2; ++nt) {
            const unsigned short* p = &Bs[cur][(wn * 32 + nt * 16 + lm) * LDS_STRIDE + bk0];
            bF[nt].q[0] = *(const uint4*)(p);
            bF[nt].q[1] = *(const uint4*)(p + 16);
        }

#pragma unroll
        for (int mt = 0; mt < 4; ++mt)
#pragma unroll
            for (int nt = 0; nt < 2; ++nt)
                acc[mt][nt] = __builtin_amdgcn_wmma_f32_16x16x32_bf16(
                    false, aF[mt].v, false, bF[nt].v,
                    (short)0, acc[mt][nt], false, false);
    }

    // Epilogue: Q0 = exp(acc * scale). C/D layout: VGPR r, lane l ->
    // (M = r + 8*(l>=16), N = l&15).
#pragma unroll
    for (int mt = 0; mt < 4; ++mt) {
#pragma unroll
        for (int nt = 0; nt < 2; ++nt) {
            int n = kBase + wn * 32 + nt * 16 + lm;
#pragma unroll
            for (int r = 0; r < 8; ++r) {
                int m = bBase + wm * 64 + mt * 16 + r + 8 * lhalf;
                Q[(size_t)m * K + n] = __expf(acc[mt][nt][r] * scale);
            }
        }
    }
}

// ---------------------------------------------------------------------------
// 4a. initial column-normalization factor: beta[b] = 1 / sum_k Q0[b,k]
__global__ __launch_bounds__(256)
void rowsum_init_kernel(const float* __restrict__ Q, float* __restrict__ beta, int K) {
    __shared__ float red[256];
    const float* row = Q + (size_t)blockIdx.x * K;
    float s = 0.0f;
    for (int k = threadIdx.x; k < K; k += 256) s += row[k];
    s = blockReduceSum(s, red);
    if (threadIdx.x == 0) beta[blockIdx.x] = 1.0f / s;
}

// 4b. zero u[K]
__global__ void zero_u_kernel(float* u, int K) {
    int i = blockIdx.x * blockDim.x + threadIdx.x;
    if (i < K) u[i] = 0.0f;
}

// 4c. column sums: u[k] += sum_b Q0[b,k] * beta[b]   (grid.y chunks of 256 b)
__global__ __launch_bounds__(256)
void colsum_kernel(const float* __restrict__ Q, const float* __restrict__ beta,
                   float* __restrict__ u, int K) {
    __shared__ float sb[256];
    const int k  = blockIdx.x * 256 + threadIdx.x;
    const int b0 = blockIdx.y * 256;
    sb[threadIdx.x] = beta[b0 + threadIdx.x];
    __syncthreads();
    float acc = 0.0f;
    const float* col = Q + (size_t)b0 * K + k;
    for (int j = 0; j < 256; ++j) acc += col[(size_t)j * K] * sb[j];
    atomicAdd(&u[k], acc);
}

// 4d. alpha[k] *= r / (alpha[k]*u_hat[k] + TINY),  r = 1/K
__global__ void alpha_upd_kernel(float* alpha, const float* u, int K, float r) {
    int k = blockIdx.x * blockDim.x + threadIdx.x;
    if (k < K) {
        float a = alpha[k];
        alpha[k] = a * r / (a * u[k] + TINY);
    }
}

// 4e. row pass: v_hat = sum_k Q0[b,k]*alpha[k]; beta *= c/(beta*v_hat + TINY)
__global__ __launch_bounds__(256)
void row_pass_kernel(const float* __restrict__ Q, const float* __restrict__ alpha,
                     float* __restrict__ beta, int K, float c) {
    __shared__ float red[256];
    const float* row = Q + (size_t)blockIdx.x * K;
    float s = 0.0f;
    for (int k = threadIdx.x; k < K; k += 256) s += row[k] * alpha[k];
    s = blockReduceSum(s, red);
    if (threadIdx.x == 0) {
        float bb = beta[blockIdx.x];
        beta[blockIdx.x] = bb * c / (bb * s + TINY);
    }
}

// ---------------------------------------------------------------------------
// 5. fused loss (one block per batch row b):
//    softcode[b,k] = qa[k]/S,  qa = Q0[b,k]*alpha[k], S = sum_k qa  (beta cancels)
//    lse = logsumexp(logits[b,:]);  out += -(1/B) * sum_k qa*(logit-lse)/S
__global__ __launch_bounds__(256)
void loss_kernel(const float* __restrict__ Q, const float* __restrict__ alpha,
                 const float* __restrict__ logits, float* __restrict__ out,
                 int K, float invB) {
    __shared__ float qa[KMAX];
    __shared__ float lg[KMAX];
    __shared__ float red[256];

    const float* qrow = Q      + (size_t)blockIdx.x * K;
    const float* lrow = logits + (size_t)blockIdx.x * K;

    float sQ = 0.0f, mx = -INFINITY;
    for (int k = threadIdx.x; k < K; k += 256) {
        float q = qrow[k] * alpha[k];
        qa[k] = q; sQ += q;
        float l = lrow[k];
        lg[k] = l; mx = fmaxf(mx, l);
    }
    sQ = blockReduceSum(sQ, red);
    mx = blockReduceMax(mx, red);

    float se = 0.0f;
    for (int k = threadIdx.x; k < K; k += 256) se += __expf(lg[k] - mx);
    se = blockReduceSum(se, red);
    float lse = mx + __logf(se);

    float dot = 0.0f;
    for (int k = threadIdx.x; k < K; k += 256) dot += qa[k] * (lg[k] - lse);
    dot = blockReduceSum(dot, red);

    if (threadIdx.x == 0) atomicAdd(out, -dot * invB / sQ);
}

// ---------------------------------------------------------------------------
extern "C" void kernel_launch(void* const* d_in, const int* in_sizes, int n_in,
                              void* d_out, int out_size, void* d_ws, size_t ws_size,
                              hipStream_t stream) {
    const float* features = (const float*)d_in[0];   // [B, D]
    const float* protos   = (const float*)d_in[1];   // [K, D]
    const float* logits   = (const float*)d_in[2];   // [B, K]
    float* out = (float*)d_out;

    // Derive dims: s0=B*D, s1=K*D, s2=B*K  ->  D = sqrt(s0*s1/s2)
    const long s0 = in_sizes[0], s1 = in_sizes[1], s2 = in_sizes[2];
    const int D = (int)llround(sqrt((double)s0 * (double)s1 / (double)s2));
    const int B = (int)(s0 / D);
    const int K = (int)(s1 / D);

    // Workspace carve-out (256B aligned): fA(bf16) fW(bf16) Q(f32) alpha beta u
    auto align256 = [](size_t x) { return (x + 255) & ~(size_t)255; };
    char* w = (char*)d_ws;
    unsigned short* fA = (unsigned short*)w; w += align256((size_t)B * D * 2);
    unsigned short* fW = (unsigned short*)w; w += align256((size_t)K * D * 2);
    float* Q     = (float*)w; w += align256((size_t)B * K * 4);
    float* alpha = (float*)w; w += align256((size_t)K * 4);
    float* beta  = (float*)w; w += align256((size_t)B * 4);
    float* u     = (float*)w; w += align256((size_t)K * 4);
    (void)ws_size; (void)n_in; (void)out_size;

    const float scale = 1.0f / (TEMPERATURE * EPSILON);   // sims/T then /EPS

    init_kernel<<<(K + 255) / 256, 256, 0, stream>>>(alpha, out, K);
    feat_cvt_kernel<<<2048, 256, 0, stream>>>(features, fA, (long)B * D);
    proto_norm_kernel<<<K, 256, 0, stream>>>(protos, fW, D);

    gemm_exp_kernel<<<dim3(K / TILE_N, B / TILE_M), 256, 0, stream>>>(
        fA, fW, Q, B, K, D, scale);

    rowsum_init_kernel<<<B, 256, 0, stream>>>(Q, beta, K);
    for (int it = 0; it < NUM_ITERS; ++it) {
        zero_u_kernel<<<(K + 255) / 256, 256, 0, stream>>>(u, K);
        colsum_kernel<<<dim3(K / 256, B / 256), 256, 0, stream>>>(Q, beta, u, K);
        alpha_upd_kernel<<<(K + 255) / 256, 256, 0, stream>>>(alpha, u, K, 1.0f / K);
        row_pass_kernel<<<B, 256, 0, stream>>>(Q, alpha, beta, K, 1.0f / B);
    }

    loss_kernel<<<B, 256, 0, stream>>>(Q, alpha, logits, out, K, 1.0f / B);
}
